// Attention_38800734552735
// MI455X (gfx1250) — matrix operand order repaired and checked by reading, not compiled, not run on hardware
//
#include <hip/hip_runtime.h>

// ---------------------------------------------------------------------------
// Fused attention block for MI455X (gfx1250, wave32, WMMA).
// All three matmul stages run on v_wmma_f32_16x16x32_bf16.
// GEMM A-tiles are staged into LDS with CDNA5 async loads (ASYNCcnt).
// ---------------------------------------------------------------------------

#define S_LEN   2048
#define HIDDEN  5120
#define NQH     40
#define NKVH    8
#define DH      128
#define QKV_DIM 7208          // (40+16)*128 + 40
#define Q_DIM   (NQH * DH)    // 5120
#define K_DIM   (NKVH * DH)   // 1024
#define EPS_F   1e-6f

typedef __bf16 bf16_t;
typedef bf16_t bf16x16 __attribute__((ext_vector_type(16)));
typedef float  v8f     __attribute__((ext_vector_type(8)));

struct alignas(16) U4 { unsigned v[4]; };
struct alignas(8)  U2 { unsigned v[2]; };

union BF16Frag {
    unsigned short us[16];
    unsigned       u32[8];
    U4             u4[2];
    bf16x16        v;
};

__device__ inline unsigned short f2bf(float f) {
    unsigned u = __float_as_uint(f);
    unsigned r = u + 0x7FFFu + ((u >> 16) & 1u);   // round-to-nearest-even
    return (unsigned short)(r >> 16);
}

__device__ inline v8f wmma_bf16(bf16x16 a, bf16x16 b, v8f c) {
    return __builtin_amdgcn_wmma_f32_16x16x32_bf16(
        /*neg_a=*/false, a, /*neg_b=*/false, b,
        /*c_mod=*/(short)0, c, /*reuse_a=*/false, /*reuse_b=*/false);
}

// CDNA5 async global->LDS copy, 2 x 16B per call. INST_OFFSET is applied to
// both the LDS and the global address (CDNA5 ISA pseudocode), so
// one LDS-base VGPR + one 64-bit global address covers both halves.
__device__ inline void async_copy_32B(void* lds_dst, const void* gsrc) {
    unsigned       ldsaddr = (unsigned)(size_t)lds_dst;       // flat LDS: offset in addr[31:0]
    unsigned long long ga  = (unsigned long long)(size_t)gsrc;
    asm volatile("global_load_async_to_lds_b128 %0, %1, off\n\t"
                 "global_load_async_to_lds_b128 %0, %1, off offset:16"
                 :: "v"(ldsaddr), "v"(ga)
                 : "memory");
}

__device__ inline void wait_asynccnt0() {
    asm volatile("s_wait_asynccnt 0x0" ::: "memory");
}

// ---------------------------------------------------------------------------
// Kernel 1: rmsnorm over HIDDEN, output bf16. One block per sequence position.
// ---------------------------------------------------------------------------
__global__ __launch_bounds__(256) void prenorm_kernel(
    const float* __restrict__ x, const float* __restrict__ w,
    unsigned short* __restrict__ out) {
    __shared__ float red[256];
    const int s = blockIdx.x;
    const float* row = x + (size_t)s * HIDDEN;
    float partial = 0.f;
    for (int i = threadIdx.x; i < HIDDEN; i += 256) {
        float v = row[i];
        partial += v * v;
    }
    red[threadIdx.x] = partial;
    __syncthreads();
    for (int off = 128; off > 0; off >>= 1) {
        if (threadIdx.x < off) red[threadIdx.x] += red[threadIdx.x + off];
        __syncthreads();
    }
    const float n = rsqrtf(red[0] * (1.f / HIDDEN) + EPS_F);
    for (int i = threadIdx.x; i < HIDDEN; i += 256)
        out[(size_t)s * HIDDEN + i] = f2bf(row[i] * n * w[i]);
}

// ---------------------------------------------------------------------------
// Kernel 2/5: C[M,N](f32) = A[M,K](bf16) @ B[K,N](f32, cast to bf16 on load).
// 256 threads = 8 waves; 128x128 tile; each wave owns a 16x128 strip (8 WMMA
// accumulators). K staged in 32-wide chunks through LDS. A-tile: async
// global->LDS (overlaps with B conversion VALU work). B-tile: coalesced f32
// loads, converted to bf16, stored transposed so B-fragments are contiguous.
// ---------------------------------------------------------------------------
__global__ __launch_bounds__(256) void gemm_bf16_kernel(
    const unsigned short* __restrict__ A, const float* __restrict__ B,
    float* __restrict__ C, int M, int N, int K) {
    __shared__ unsigned short At[128][48];   // [m][k], padded stride (96B rows)
    __shared__ unsigned short Bt[128][48];   // [n][k] transposed, padded

    const int tid  = threadIdx.x;
    const int wave = tid >> 5;
    const int lane = tid & 31;
    const int lhi  = lane >> 4;
    const int llo  = lane & 15;
    const int m0   = blockIdx.y * 128;
    const int n0   = blockIdx.x * 128;

    const v8f vzero = {0.f, 0.f, 0.f, 0.f, 0.f, 0.f, 0.f, 0.f};
    v8f acc[8];
    for (int i = 0; i < 8; ++i) acc[i] = vzero;

    for (int k0 = 0; k0 < K; k0 += 32) {
        // ---- stage A tile: 128 rows x 32 bf16, async global->LDS ----
        {
            const int r  = tid >> 1;
            const int cb = (tid & 1) * 16;
            const unsigned short* src = A + (size_t)(m0 + r) * K + (k0 + cb);
            async_copy_32B(&At[r][cb], src);
        }
        // ---- stage B tile transposed: coalesced f32 loads, cvt to bf16 ----
        {
            const int col = tid & 127;
            const int kc  = (tid >> 7) * 16;
            const int n   = n0 + col;
            BF16Frag fr;
            if (n < N) {
                const float* bs = B + (size_t)(k0 + kc) * N + n;
                for (int i = 0; i < 16; ++i) fr.us[i] = f2bf(bs[(size_t)i * N]);
            } else {
                for (int i = 0; i < 16; ++i) fr.us[i] = 0;
            }
            *(U4*)&Bt[col][kc]     = fr.u4[0];
            *(U4*)&Bt[col][kc + 8] = fr.u4[1];
        }
        if (k0 + 32 < K) {  // CDNA5 global_prefetch_b8 for next chunk
            __builtin_prefetch(A + (size_t)(m0 + (tid >> 1)) * K + (k0 + 32), 0, 0);
            __builtin_prefetch(B + (size_t)(k0 + 32 + (tid >> 7) * 16) * N +
                                   (n0 + (tid & 127)), 0, 0);
        }
        wait_asynccnt0();   // drain ASYNCcnt before signaling the barrier
        __syncthreads();

        // ---- A fragment (16x32): lane<16 holds K{0..7,16..23}, else +8 ----
        BF16Frag af;
        {
            const int arow = wave * 16 + llo;
            for (int j2 = 0; j2 < 4; ++j2) {
                af.u32[j2]     = *(const unsigned*)&At[arow][8 * lhi + 2 * j2];
                af.u32[4 + j2] = *(const unsigned*)&At[arow][16 + 8 * lhi + 2 * j2];
            }
        }
        // ---- 8 WMMAs across the 128-wide N strip ----
        for (int sub = 0; sub < 8; ++sub) {
            BF16Frag bfm;
            const int bcol = sub * 16 + llo;    // B column = lane&15
            bfm.u4[0] = *(const U4*)&Bt[bcol][8 * lhi];      // K rows 8*lhi..+7, 16+8*lhi..+7
            bfm.u4[1] = *(const U4*)&Bt[bcol][16 + 8 * lhi];
            acc[sub] = wmma_bf16(af.v, bfm.v, acc[sub]);
        }
        __syncthreads();
    }

    // ---- store: C/D layout -> M = j + 8*lhi, N = lane&15 ----
    for (int sub = 0; sub < 8; ++sub) {
        const int n = n0 + sub * 16 + llo;
        if (n < N) {
            for (int j = 0; j < 8; ++j) {
                const int row = m0 + wave * 16 + j + 8 * lhi;
                C[(size_t)row * N + n] = acc[sub][j];
            }
        }
    }
}

// ---------------------------------------------------------------------------
// Kernel 3: split qkv; per-head rmsnorm + RoPE for q/k (scale folded into q);
// V stored transposed [kvh][D][S]; sigmoid gate. One block per position,
// one wave per head-unit (lane owns 4 contiguous d's; RoPE partner via xor16).
// ---------------------------------------------------------------------------
__global__ __launch_bounds__(256) void qkv_prep_kernel(
    const float* __restrict__ qkv, const float* __restrict__ cosb,
    const float* __restrict__ sinb, const float* __restrict__ wqn,
    const float* __restrict__ wkn, unsigned short* __restrict__ qh,
    unsigned short* __restrict__ kh, unsigned short* __restrict__ vT,
    float* __restrict__ gate) {
    const int s = blockIdx.x;
    const float* row = qkv + (size_t)s * QKV_DIM;
    const int tid = threadIdx.x, wave = tid >> 5, lane = tid & 31;

    if (tid < NQH) {
        const float g = row[Q_DIM + 2 * K_DIM + tid];
        gate[(size_t)s * NQH + tid] = 1.f / (1.f + __expf(-g));
    }

    const int d0 = lane * 4;
    const float4 cs4 = *(const float4*)(cosb + (size_t)s * DH + d0);
    const float4 sn4 = *(const float4*)(sinb + (size_t)s * DH + d0);
    const float qscale = 0.08838834764831845f;  // 1/sqrt(128)

    for (int u = wave; u < NQH + 2 * NKVH; u += 8) {
        if (u < NQH + NKVH) {
            const bool isq = (u < NQH);
            const int h = isq ? u : u - NQH;
            const float* base = row + (isq ? h * DH : Q_DIM + h * DH);
            const float4 xv = *(const float4*)(base + d0);
            float ss = xv.x * xv.x + xv.y * xv.y + xv.z * xv.z + xv.w * xv.w;
            for (int m = 16; m >= 1; m >>= 1) ss += __shfl_xor(ss, m, 32);
            const float n = rsqrtf(ss * (1.f / DH) + EPS_F);
            const float* wn = isq ? wqn : wkn;
            float y0 = xv.x * n * wn[d0 + 0];
            float y1 = xv.y * n * wn[d0 + 1];
            float y2 = xv.z * n * wn[d0 + 2];
            float y3 = xv.w * n * wn[d0 + 3];
            // RoPE: partner element at d +/- 64 lives in lane^16
            float o0 = __shfl_xor(y0, 16, 32);
            float o1 = __shfl_xor(y1, 16, 32);
            float o2 = __shfl_xor(y2, 16, 32);
            float o3 = __shfl_xor(y3, 16, 32);
            const float sgn = (lane < 16) ? -1.f : 1.f;
            float z0 = y0 * cs4.x + sgn * o0 * sn4.x;
            float z1 = y1 * cs4.y + sgn * o1 * sn4.y;
            float z2 = y2 * cs4.z + sgn * o2 * sn4.z;
            float z3 = y3 * cs4.w + sgn * o3 * sn4.w;
            if (isq) { z0 *= qscale; z1 *= qscale; z2 *= qscale; z3 *= qscale; }
            union { unsigned short us[4]; U2 u; } p;
            p.us[0] = f2bf(z0); p.us[1] = f2bf(z1);
            p.us[2] = f2bf(z2); p.us[3] = f2bf(z3);
            unsigned short* dst = (isq ? qh : kh) + ((size_t)h * S_LEN + s) * DH + d0;
            *(U2*)dst = p.u;
        } else {
            const int h = u - NQH - NKVH;
            const float* base = row + Q_DIM + K_DIM + h * DH;
            const float4 xv = *(const float4*)(base + d0);
            unsigned short* dst = vT + (size_t)h * DH * S_LEN + s;
            dst[(size_t)(d0 + 0) * S_LEN] = f2bf(xv.x);
            dst[(size_t)(d0 + 1) * S_LEN] = f2bf(xv.y);
            dst[(size_t)(d0 + 2) * S_LEN] = f2bf(xv.z);
            dst[(size_t)(d0 + 3) * S_LEN] = f2bf(xv.w);
        }
    }
}

// ---------------------------------------------------------------------------
// Kernel 4: flash attention. One wave per (head, 16-query block). Online
// softmax over 32-key chunks; QK^T and PV both on WMMA bf16. P re-striped
// from C-layout to A-layout through 1KB of LDS. GQA: kv head = h/5.
// ---------------------------------------------------------------------------
__global__ __launch_bounds__(32) void attn_kernel(
    const unsigned short* __restrict__ qb, const unsigned short* __restrict__ kb,
    const unsigned short* __restrict__ vtb, const float* __restrict__ gate,
    unsigned short* __restrict__ ob) {
    __shared__ unsigned short Plds[16][32];
    const int lane = threadIdx.x;
    const int lhi = lane >> 4, llo = lane & 15;
    const int q0 = blockIdx.x * 16;
    const int h  = blockIdx.y;
    const int kvh = h / (NQH / NKVH);
    const unsigned short* Q  = qb  + (size_t)h   * S_LEN * DH;
    const unsigned short* Kp = kb  + (size_t)kvh * S_LEN * DH;
    const unsigned short* VT = vtb + (size_t)kvh * DH * S_LEN;

    // Q A-fragments for all of D=128 (4 frags of 16x32), loaded once.
    BF16Frag aq[4];
    {
        const unsigned short* qrow = Q + (size_t)(q0 + llo) * DH;
        for (int c = 0; c < 4; ++c) {
            aq[c].u4[0] = *(const U4*)(qrow + 32 * c + 16 * lhi);
            aq[c].u4[1] = *(const U4*)(qrow + 32 * c + 16 * lhi + 8);
        }
    }

    const v8f vzero = {0.f, 0.f, 0.f, 0.f, 0.f, 0.f, 0.f, 0.f};
    v8f out_acc[8];
    float m_run[8], l_run[8];
    for (int i = 0; i < 8; ++i) { out_acc[i] = vzero; m_run[i] = -3.0e38f; l_run[i] = 0.f; }

    for (int kk = 0; kk < S_LEN; kk += 32) {
        // ---- scores: two 16x16 tiles over 32 keys, K-dim = D in 4 chunks ----
        v8f s0 = vzero, s1 = vzero;
        for (int c = 0; c < 4; ++c) {
            BF16Frag b0, b1;
            const unsigned short* kr0 = Kp + (size_t)(kk + llo) * DH + 32 * c + 16 * lhi;
            const unsigned short* kr1 = Kp + (size_t)(kk + 16 + llo) * DH + 32 * c + 16 * lhi;
            b0.u4[0] = *(const U4*)(kr0); b0.u4[1] = *(const U4*)(kr0 + 8);
            b1.u4[0] = *(const U4*)(kr1); b1.u4[1] = *(const U4*)(kr1 + 8);
            s0 = wmma_bf16(aq[c].v, b0.v, s0);
            s1 = wmma_bf16(aq[c].v, b1.v, s1);
        }
        // ---- online softmax; rows = vgpr idx (+8 for hi lanes), cols = lanes ----
        float alpha[8];
        for (int j = 0; j < 8; ++j) {
            float mc = fmaxf(s0[j], s1[j]);
            mc = fmaxf(mc, __shfl_xor(mc, 1, 32));
            mc = fmaxf(mc, __shfl_xor(mc, 2, 32));
            mc = fmaxf(mc, __shfl_xor(mc, 4, 32));
            mc = fmaxf(mc, __shfl_xor(mc, 8, 32));
            const float mn = fmaxf(m_run[j], mc);
            alpha[j] = __expf(m_run[j] - mn);
            const float p0 = __expf(s0[j] - mn);
            const float p1 = __expf(s1[j] - mn);
            float rs = p0 + p1;
            rs += __shfl_xor(rs, 1, 32);
            rs += __shfl_xor(rs, 2, 32);
            rs += __shfl_xor(rs, 4, 32);
            rs += __shfl_xor(rs, 8, 32);
            l_run[j] = l_run[j] * alpha[j] + rs;
            m_run[j] = mn;
            Plds[j + 8 * lhi][llo]      = f2bf(p0);
            Plds[j + 8 * lhi][llo + 16] = f2bf(p1);
        }
        for (int sub = 0; sub < 8; ++sub)
            for (int j = 0; j < 8; ++j) out_acc[sub][j] *= alpha[j];
        __syncthreads();
        // ---- restripe P (C-layout) into A-fragment (16x32) ----
        BF16Frag pa;
        for (int j2 = 0; j2 < 4; ++j2) {
            pa.u32[j2]     = *(const unsigned*)&Plds[llo][8 * lhi + 2 * j2];
            pa.u32[4 + j2] = *(const unsigned*)&Plds[llo][16 + 8 * lhi + 2 * j2];
        }
        // ---- out += P @ V : 8 d-subtiles, V^T gives contiguous B frags ----
        for (int sub = 0; sub < 8; ++sub) {
            BF16Frag bv;
            const unsigned short* vr = VT + (size_t)(sub * 16 + llo) * S_LEN + kk + 8 * lhi;
            bv.u4[0] = *(const U4*)(vr);
            bv.u4[1] = *(const U4*)(vr + 16);
            out_acc[sub] = wmma_bf16(pa.v, bv.v, out_acc[sub]);
        }
        __syncthreads();
    }

    // ---- finalize: /l, *gate, store bf16 [S][NQ*D] ----
    float rowscale[8];
    for (int j = 0; j < 8; ++j) {
        const int row = q0 + j + 8 * lhi;
        rowscale[j] = gate[(size_t)row * NQH + h] / l_run[j];
    }
    for (int sub = 0; sub < 8; ++sub) {
        for (int j = 0; j < 8; ++j) {
            const int row = q0 + j + 8 * lhi;
            const int col = h * DH + sub * 16 + llo;
            ob[(size_t)row * (NQH * DH) + col] = f2bf(out_acc[sub][j] * rowscale[j]);
        }
    }
}

// ---------------------------------------------------------------------------
// Launcher
// ---------------------------------------------------------------------------
extern "C" void kernel_launch(void* const* d_in, const int* in_sizes, int n_in,
                              void* d_out, int out_size, void* d_ws, size_t ws_size,
                              hipStream_t stream) {
    (void)in_sizes; (void)n_in; (void)out_size; (void)ws_size;
    const float* x        = (const float*)d_in[0];
    const float* rope_cos = (const float*)d_in[1];
    const float* rope_sin = (const float*)d_in[2];
    const float* w_pre    = (const float*)d_in[3];
    const float* w_qkv    = (const float*)d_in[4];
    const float* w_qn     = (const float*)d_in[5];
    const float* w_kn     = (const float*)d_in[6];
    const float* w_proj   = (const float*)d_in[7];
    float* out = (float*)d_out;

    char* ws = (char*)d_ws;
    size_t off = 0;
    auto carve = [&](size_t bytes) -> void* {
        void* p = ws + off;
        off += (bytes + 255) & ~(size_t)255;
        return p;
    };
    unsigned short* normed = (unsigned short*)carve((size_t)S_LEN * HIDDEN * 2);
    float*          qkv    = (float*)         carve((size_t)S_LEN * QKV_DIM * 4);
    unsigned short* qh     = (unsigned short*)carve((size_t)NQH  * S_LEN * DH * 2);
    unsigned short* kh     = (unsigned short*)carve((size_t)NKVH * S_LEN * DH * 2);
    unsigned short* vT     = (unsigned short*)carve((size_t)NKVH * DH * S_LEN * 2);
    float*          gate   = (float*)         carve((size_t)S_LEN * NQH * 4);
    unsigned short* aout   = (unsigned short*)carve((size_t)S_LEN * NQH * DH * 2);

    prenorm_kernel<<<S_LEN, 256, 0, stream>>>(x, w_pre, normed);

    dim3 g1((QKV_DIM + 127) / 128, S_LEN / 128);   // 57 x 16
    gemm_bf16_kernel<<<g1, 256, 0, stream>>>(normed, w_qkv, qkv,
                                             S_LEN, QKV_DIM, HIDDEN);

    qkv_prep_kernel<<<S_LEN, 256, 0, stream>>>(qkv, rope_cos, rope_sin,
                                               w_qn, w_kn, qh, kh, vT, gate);

    attn_kernel<<<dim3(S_LEN / 16, NQH), 32, 0, stream>>>(qh, kh, vT, gate, aout);

    dim3 g2((HIDDEN + 127) / 128, S_LEN / 128);    // 40 x 16
    gemm_bf16_kernel<<<g2, 256, 0, stream>>>(aout, w_proj, out,
                                             S_LEN, HIDDEN, NQH * DH);
}
